// CIN_56298431315971
// MI455X (gfx1250) — matrix operand order repaired and checked by reading, not compiled, not run on hardware
//
#include <hip/hip_runtime.h>

// ---------------------------------------------------------------------------
// CIN (xDeepFM) for MI455X / gfx1250: bf16 WMMA (V_WMMA_F32_16X16X32_BF16)
// One block (4 wave32 waves) per batch sample. Per layer: M=128 (k), N=32 (d),
// K=G*64 GEMM; B matrix (outer products xi[g,d]*x0[h,d]) built on the fly.
// Each wave owns 2 M-tiles x 2 N-tiles (4 pk_mul per WMMA). The g-loop body
// covers 2 g's with all four B fragments live before the 8 WMMAs, so RA must
// keep them in distinct register blocks and the WMMA->VALU WAR window (4
// coexec slots for *_BF16) is filled by the next body's independent pk_muls.
// sched_group_barrier pins the [VALU*32][WMMA*8] shape per body.
// ---------------------------------------------------------------------------

typedef __attribute__((ext_vector_type(16))) __bf16 bf16x16;
typedef __attribute__((ext_vector_type(8)))  __bf16 bf16x8;
typedef __attribute__((ext_vector_type(8)))  float  f32x8;
typedef __attribute__((ext_vector_type(4)))  float  f32x4;

#define S0 72    // LDS row stride (bf16 elems) for x0 transpose  (>=64, rows 16B-aligned)
#define S1 136   // LDS row stride for 128-wide layer buffers     (>=128, rows 16B-aligned)

#define WMMA_BF16(A, B, C) \
  __builtin_amdgcn_wmma_f32_16x16x32_bf16(false, (A), false, (B), (short)0, (C), false, false)

#if __has_builtin(__builtin_amdgcn_sched_group_barrier)
#define SGB(mask, cnt, id) __builtin_amdgcn_sched_group_barrier((mask), (cnt), (id))
#else
#define SGB(mask, cnt, id)
#endif

__device__ __forceinline__ bf16x16 cat16(bf16x8 lo, bf16x8 hi) {
  bf16x16 r;
#pragma unroll
  for (int i = 0; i < 8; ++i) { r[i] = lo[i]; r[i + 8] = hi[i]; }
  return r;
}

__device__ __forceinline__ bf16x16 ldA(const __bf16* p) {
  return cat16(*(const bf16x8*)p, *(const bf16x8*)(p + 16));
}

// One CIN layer: out_t[d][k] = relu( sum_{g,h} xi[g,d]*x0[h,d]*W[k, g*64+h] + bias[k] )
// xi_t / x0t / out_t live in LDS, transposed: row = d, col = feature.
// W is bf16 row-major [128][G*64] in global (L2 / WGP$-resident).
template<int G, int SIN>
__device__ __forceinline__ void cin_layer(
    const __bf16* xi_t, const __bf16* x0t,
    const __bf16* __restrict__ W, const float* __restrict__ bias,
    __bf16* out_t, float* __restrict__ pooled,
    int lane, int wave, int tid)
{
  const int l15  = lane & 15;
  const int klo  = (lane < 16) ? 0 : 8;    // A-operand K split: {0..7,16..23} / {8..15,24..31}
  const int koff = (lane < 16) ? 0 : 8;    // B-operand K split: {0..7,16..23} / {8..15,24..31}
  const int mt0  = wave * 2;               // first of two M-tiles (waves 0..3 -> M 0..7)
  const int d0   = l15;                    // N (=d) column, N-tile 0
  const int d1   = 16 + l15;               // N (=d) column, N-tile 1
  const int row0 = mt0 * 16 + l15;         // A (=k) row of first M-tile

  const int ROWSTEP = 16 * (G * 64);       // element offset between the 2 M-tiles

  f32x8 acc00 = {0.f,0.f,0.f,0.f,0.f,0.f,0.f,0.f};  // (mt0, nt0)
  f32x8 acc10 = acc00;                                // (mt1, nt0)
  f32x8 acc01 = acc00;                                // (mt0, nt1)
  f32x8 acc11 = acc00;                                // (mt1, nt1)

  const __bf16* W0 = W + (size_t)row0 * (G * 64) + klo;

#pragma unroll
  for (int hb = 0; hb < 2; ++hb) {         // two 32-wide h-blocks per g
    // x0 chunks for this lane's two d columns: two runs of 8 h values, 16 apart (B layout)
    const __bf16* xp0 = x0t + d0 * S0 + hb * 32 + koff;
    const __bf16* xp1 = x0t + d1 * S0 + hb * 32 + koff;
    bf16x16 xv0 = cat16(*(const bf16x8*)xp0, *(const bf16x8*)(xp0 + 16));
    bf16x16 xv1 = cat16(*(const bf16x8*)xp1, *(const bf16x8*)(xp1 + 16));

    const __bf16* ap  = W0 + hb * 32;      // advances by 128 elems per body (2 g's)
    const __bf16* xr0 = xi_t + d0 * SIN;
    const __bf16* xr1 = xi_t + d1 * SIN;

#pragma unroll 1
    for (int g = 0; g < G; g += 2) {
      __bf16 s0a = xr0[g];
      __bf16 s1a = xr1[g];
      __bf16 s0b = xr0[g + 1];
      __bf16 s1b = xr1[g + 1];
      // Four B fragments, all live before the WMMA block -> distinct registers
      bf16x16 bf0a = xv0 * s0a;
      bf16x16 bf1a = xv1 * s1a;
      bf16x16 bf0b = xv0 * s0b;
      bf16x16 bf1b = xv1 * s1b;
      // Four A fragments (2 M-tiles x 2 g's), 16-bit A layout
      bf16x16 a0a = ldA(ap);
      bf16x16 a1a = ldA(ap + ROWSTEP);
      bf16x16 a0b = ldA(ap + 64);
      bf16x16 a1b = ldA(ap + ROWSTEP + 64);
      // g
      acc00 = WMMA_BF16(a0a, bf0a, acc00);
      acc10 = WMMA_BF16(a1a, bf0a, acc10);
      acc01 = WMMA_BF16(a0a, bf1a, acc01);
      acc11 = WMMA_BF16(a1a, bf1a, acc11);
      // g + 1
      acc00 = WMMA_BF16(a0b, bf0b, acc00);
      acc10 = WMMA_BF16(a1b, bf0b, acc10);
      acc01 = WMMA_BF16(a0b, bf1b, acc01);
      acc11 = WMMA_BF16(a1b, bf1b, acc11);
      ap += 128;
      // Pin schedule shape: fragment build first, then the 8 WMMAs.
      SGB(0x00000002 /*VALU*/, 32, 0);
      SGB(0x00000008 /*MFMA-WMMA*/, 8, 0);
    }
  }

  // Epilogue: bias + relu, pack to bf16, store transposed (out_t[d][k]).
  // C/D layout: VGPR r -> row r (lanes 0-15) or row 8+r (lanes 16-31); N = lane&15.
  const int kofs0 = mt0 * 16 + klo;        // 8 consecutive k values per lane, tile 0
  const int kofs1 = kofs0 + 16;            // tile 1
  f32x4 bA0 = *(const f32x4*)(bias + kofs0);
  f32x4 bB0 = *(const f32x4*)(bias + kofs0 + 4);
  f32x4 bA1 = *(const f32x4*)(bias + kofs1);
  f32x4 bB1 = *(const f32x4*)(bias + kofs1 + 4);
  float bb0[8] = {bA0.x, bA0.y, bA0.z, bA0.w, bB0.x, bB0.y, bB0.z, bB0.w};
  float bb1[8] = {bA1.x, bA1.y, bA1.z, bA1.w, bB1.x, bB1.y, bB1.z, bB1.w};

  bf16x8 pk00, pk10, pk01, pk11;
#pragma unroll
  for (int r = 0; r < 8; ++r) {
    float v00 = acc00[r] + bb0[r]; v00 = v00 > 0.f ? v00 : 0.f;
    float v10 = acc10[r] + bb1[r]; v10 = v10 > 0.f ? v10 : 0.f;
    float v01 = acc01[r] + bb0[r]; v01 = v01 > 0.f ? v01 : 0.f;
    float v11 = acc11[r] + bb1[r]; v11 = v11 > 0.f ? v11 : 0.f;
    pk00[r] = (__bf16)v00;
    pk10[r] = (__bf16)v10;
    pk01[r] = (__bf16)v01;
    pk11[r] = (__bf16)v11;
  }
  *(bf16x8*)(out_t + d0 * S1 + kofs0) = pk00;
  *(bf16x8*)(out_t + d0 * S1 + kofs1) = pk10;
  *(bf16x8*)(out_t + d1 * S1 + kofs0) = pk01;
  *(bf16x8*)(out_t + d1 * S1 + kofs1) = pk11;
  __syncthreads();

  // pooled[k] = sum_d fm[k,d]   (block has exactly 128 threads)
  {
    float s = 0.f;
#pragma unroll
    for (int dd = 0; dd < 32; ++dd) s += (float)out_t[dd * S1 + tid];
    pooled[tid] = s;
  }
  __syncthreads();
}

__global__ __launch_bounds__(128) void cin_main(
    const float* __restrict__ x,
    const __bf16* __restrict__ w0, const __bf16* __restrict__ w1,
    const __bf16* __restrict__ w2,
    const float* __restrict__ b0, const float* __restrict__ b1,
    const float* __restrict__ b2,
    float* __restrict__ out)
{
  __shared__ __bf16 xt0[32 * S0];   // x0 transposed: [d][h], bf16
  __shared__ __bf16 bufA[32 * S1];  // layer fm buffers, transposed [d][k]
  __shared__ __bf16 bufB[32 * S1];

  const int tid  = threadIdx.x;
  const int lane = tid & 31;
  const int wave = tid >> 5;
  const size_t b = blockIdx.x;

  // Stage + convert + transpose x[b] (64x32 f32) -> xt0[d][h] bf16
  const float* xb = x + b * (64 * 32);
#pragma unroll
  for (int i = 0; i < 16; ++i) {
    int idx = tid + 128 * i;           // idx = h*32 + d
    int h = idx >> 5, dd = idx & 31;
    xt0[dd * S0 + h] = (__bf16)xb[idx];
  }
  __syncthreads();

  float* ob = out + b * 384;
  cin_layer<64,  S0>(xt0,  xt0, w0, b0, bufA, ob,       lane, wave, tid);
  cin_layer<128, S1>(bufA, xt0, w1, b1, bufB, ob + 128, lane, wave, tid);
  cin_layer<128, S1>(bufB, xt0, w2, b2, bufA, ob + 256, lane, wave, tid);
}

// Elementwise f32 -> bf16 weight conversion into workspace (weights are
// (k, g, h) row-major, which is exactly the [128][G*64] layout the GEMM wants).
__global__ __launch_bounds__(256) void cvt_bf16(const float* __restrict__ src,
                                                __bf16* __restrict__ dst, int n) {
  int i = blockIdx.x * 256 + threadIdx.x;
  if (i < n) dst[i] = (__bf16)src[i];
}

extern "C" void kernel_launch(void* const* d_in, const int* in_sizes, int n_in,
                              void* d_out, int out_size, void* d_ws, size_t ws_size,
                              hipStream_t stream) {
  const float* x  = (const float*)d_in[0];
  const float* k0 = (const float*)d_in[1];
  const float* k1 = (const float*)d_in[2];
  const float* k2 = (const float*)d_in[3];
  const float* b0 = (const float*)d_in[4];
  const float* b1 = (const float*)d_in[5];
  const float* b2 = (const float*)d_in[6];

  const int n0 = 128 * 64 * 64;    // 524288
  const int n1 = 128 * 128 * 64;   // 1048576
  __bf16* w0 = (__bf16*)d_ws;
  __bf16* w1 = w0 + n0;
  __bf16* w2 = w1 + n1;

  cvt_bf16<<<n0 / 256, 256, 0, stream>>>(k0, w0, n0);
  cvt_bf16<<<n1 / 256, 256, 0, stream>>>(k1, w1, n1);
  cvt_bf16<<<n1 / 256, 256, 0, stream>>>(k2, w2, n1);

  cin_main<<<8192, 128, 0, stream>>>(x, w0, w1, w2, b0, b1, b2, (float*)d_out);
}
